// multi_head_attention_32229434589370
// MI455X (gfx1250) — matrix operand order repaired and checked by reading, not compiled, not run on hardware
//
#include <hip/hip_runtime.h>
#include <hip/hip_bf16.h>

// ---------------------------------------------------------------------------
// MI455X (gfx1250) fused multi-head attention, bf16 WMMA path.
//   K1: weight fp32 -> bf16 conversion
//   K2: Q/K/V projections (X @ W^T + b), bf16 out; V written transposed
//   K3: flash-style attention (QK^T -> softmax(masked, j<=i+1) -> PV)
//   K4: output projection (ctx @ Wc^T + bc) -> fp32
// All GEMM fragments loaded straight from global in contiguous k order (k = 16*(lane/16) + s)
// (192 MB L2 makes LDS tiling unnecessary at these sizes).
// ---------------------------------------------------------------------------

typedef __attribute__((ext_vector_type(16))) __bf16 v16bf;
typedef __attribute__((ext_vector_type(8)))  __bf16 v8bf;
typedef __attribute__((ext_vector_type(8)))  float  v8f;
typedef __attribute__((ext_vector_type(4)))  float  v4f;

#define D_MODEL  512
#define N_HEAD   8
#define HEAD_DIM 64
#define B_SZ     8
#define T_SEQ    2048
#define M_TOT    (B_SZ * T_SEQ)   // 16384 rows

__device__ __forceinline__ v8f wmma_bf16(v16bf a, v16bf b, v8f c) {
  // (neg_a, A, neg_b, B, c_mod, C, reuse_a, reuse_b)
  return __builtin_amdgcn_wmma_f32_16x16x32_bf16(false, a, false, b,
                                                 (short)0, c, false, false);
}

// A-fragment (16x32, M=lane&15) from a bf16 row: two contiguous 16B chunks.
// lane<16: k = 0..15; lane>=16 (h8=8): k = 16..31 (contiguous order).
__device__ __forceinline__ v16bf load_afrag_bf16(const __bf16* row, int h8) {
  v8bf lo = *(const v8bf*)(row + 2 * h8);
  v8bf hi = *(const v8bf*)(row + 2 * h8 + 8);
  v16bf r;
#pragma unroll
  for (int i = 0; i < 8; ++i) { r[i] = lo[i]; r[i + 8] = hi[i]; }
  return r;
}

// Same A-fragment but sourced from fp32 with in-register cvt to bf16.
__device__ __forceinline__ v16bf load_afrag_f32(const float* row, int h8) {
  v4f x0 = *(const v4f*)(row + 2 * h8);
  v4f x1 = *(const v4f*)(row + 2 * h8 + 4);
  v4f x2 = *(const v4f*)(row + 2 * h8 + 8);
  v4f x3 = *(const v4f*)(row + 2 * h8 + 12);
  v16bf r;
#pragma unroll
  for (int i = 0; i < 4; ++i) {
    r[i]      = (__bf16)x0[i];
    r[i + 4]  = (__bf16)x1[i];
    r[i + 8]  = (__bf16)x2[i];
    r[i + 12] = (__bf16)x3[i];
  }
  return r;
}

// ---------------------------------------------------------------------------
// K1: convert the four 512x512 fp32 weights to bf16 (contiguous in ws).
// ---------------------------------------------------------------------------
__global__ void wcvt_kernel(const float* __restrict__ w0, const float* __restrict__ w1,
                            const float* __restrict__ w2, const float* __restrict__ w3,
                            __bf16* __restrict__ out) {
  const int n = 4 * D_MODEL * D_MODEL;
  for (int idx = blockIdx.x * blockDim.x + threadIdx.x; idx < n;
       idx += gridDim.x * blockDim.x) {
    int which = idx >> 18;            // 512*512 = 2^18
    int off   = idx & 0x3FFFF;
    const float* src = (which == 0) ? w0 : (which == 1) ? w1 : (which == 2) ? w2 : w3;
    out[idx] = (__bf16)src[off];
  }
}

// ---------------------------------------------------------------------------
// K2: Y = X @ W^T + b.  z = 0/1/2 selects q/k/v.
// 256 threads = 8 waves (4x2), WG tile 128x64, wave tile 32x32 (2x2 C-frags).
// q,k -> [BH, T, 64] row-major bf16;  v -> [BH, 64, T] transposed bf16.
// ---------------------------------------------------------------------------
__global__ __launch_bounds__(256) void proj_qkv_kernel(
    const float* __restrict__ xq, const float* __restrict__ xk,
    const float* __restrict__ xv, const __bf16* __restrict__ wbf,
    const float* __restrict__ bq, const float* __restrict__ bk,
    const float* __restrict__ bv,
    __bf16* __restrict__ qh, __bf16* __restrict__ kh, __bf16* __restrict__ vT) {
  const int z = blockIdx.z;
  const float*  x    = (z == 0) ? xq : (z == 1) ? xk : xv;
  const __bf16* W    = wbf + (size_t)z * D_MODEL * D_MODEL;
  const float*  bias = (z == 0) ? bq : (z == 1) ? bk : bv;

  const int lane   = threadIdx.x & 31;
  const int wave   = threadIdx.x >> 5;
  const int lane15 = lane & 15;
  const int h8  = (lane & 16) ? 8 : 0;
  const int h16 = (lane & 16) ? 16 : 0;

  const int m_wave = blockIdx.x * 128 + (wave >> 1) * 32;
  const int n_wave = blockIdx.y * 64  + (wave & 1) * 32;

  v8f c[2][2] = {{{}, {}}, {{}, {}}};

  for (int k0 = 0; k0 < D_MODEL; k0 += 32) {
    v16bf a[2], b[2];
#pragma unroll
    for (int mi = 0; mi < 2; ++mi)
      a[mi] = load_afrag_f32(x + (size_t)(m_wave + mi * 16 + lane15) * D_MODEL + k0, h8);
#pragma unroll
    for (int ni = 0; ni < 2; ++ni)
      b[ni] = *(const v16bf*)(W + (size_t)(n_wave + ni * 16 + lane15) * D_MODEL + k0 + h16);
#pragma unroll
    for (int mi = 0; mi < 2; ++mi)
#pragma unroll
      for (int ni = 0; ni < 2; ++ni)
        c[mi][ni] = wmma_bf16(a[mi], b[ni], c[mi][ni]);
  }

#pragma unroll
  for (int ni = 0; ni < 2; ++ni) {
    const int   n  = n_wave + ni * 16 + lane15;
    const float bb = bias[n];
    const int   h  = n >> 6, d = n & 63;
#pragma unroll
    for (int mi = 0; mi < 2; ++mi) {
      const int mbase = m_wave + mi * 16 + h8;       // 8 consecutive rows
      const int b_    = mbase >> 11;                 // batch (T=2048 aligned)
      if (z == 2) {
        // transposed V store: vT[((b*8+h)*64 + d)*2048 + t], 8 contig bf16
        const int t0 = mbase & 2047;
        v8bf pk;
#pragma unroll
        for (int r = 0; r < 8; ++r) pk[r] = (__bf16)(c[mi][ni][r] + bb);
        *(v8bf*)(vT + ((size_t)(b_ * N_HEAD + h) * HEAD_DIM + d) * T_SEQ + t0) = pk;
      } else {
        __bf16* dst = (z == 0) ? qh : kh;
#pragma unroll
        for (int r = 0; r < 8; ++r) {
          const int m = mbase + r, t = m & 2047;
          dst[((size_t)(b_ * N_HEAD + h) * T_SEQ + t) * HEAD_DIM + d] =
              (__bf16)(c[mi][ni][r] + bb);
        }
      }
    }
  }
}

// ---------------------------------------------------------------------------
// K3: flash attention.  grid = (T/128, B*H); 8 waves, 16 query rows per wave.
// Row stats reduced across half-wave (rows live in one 16-lane half).
// Causal quirk: key j visible iff j <= i+1.
// ---------------------------------------------------------------------------
__global__ __launch_bounds__(256) void attn_kernel(
    const __bf16* __restrict__ qh, const __bf16* __restrict__ kh,
    const __bf16* __restrict__ vT, __bf16* __restrict__ ctx) {
  __shared__ __align__(32) __bf16 pstage[8][512];   // 1KB P-swizzle per wave

  const int lane   = threadIdx.x & 31;
  const int wave   = threadIdx.x >> 5;
  const int lane15 = lane & 15;
  const int h8  = (lane & 16) ? 8 : 0;
  const int h16 = (lane & 16) ? 16 : 0;

  const int bh = blockIdx.y;               // 0..63
  const int b  = bh >> 3, h = bh & 7;
  const int i0 = blockIdx.x * 128 + wave * 16;

  const __bf16* Q = qh + (size_t)bh * T_SEQ * HEAD_DIM;
  const __bf16* K = kh + (size_t)bh * T_SEQ * HEAD_DIM;
  const __bf16* V = vT + (size_t)bh * HEAD_DIM * T_SEQ;

  // Q A-fragments for the whole key sweep (d = 0..31 and 32..63).
  const v16bf aq0 = load_afrag_bf16(Q + (size_t)(i0 + lane15) * HEAD_DIM,      h8);
  const v16bf aq1 = load_afrag_bf16(Q + (size_t)(i0 + lane15) * HEAD_DIM + 32, h8);

  v8f o[4] = {{}, {}, {}, {}};
  float mrow[8], lrow[8];
#pragma unroll
  for (int r = 0; r < 8; ++r) { mrow[r] = -3.0e38f; lrow[r] = 0.0f; }

  __bf16* pst = &pstage[wave][0];
  const float scale = 0.125f;              // 1/sqrt(64)

  const int jend  = i0 + 16;               // max visible key for this row block
  const int jstop = (jend < T_SEQ - 1) ? jend : (T_SEQ - 1);

  for (int j0 = 0; j0 <= jstop; j0 += 32) {
    if (j0 + 32 < T_SEQ) {                 // warm L2/WGP$ for next key block
      __builtin_prefetch(K + (size_t)(j0 + 32 + lane15) * HEAD_DIM, 0, 0);
    }
    // S = Q @ K^T for 32 keys (two 16-key C-fragments)
    v8f z0 = {}, s0, s1;
    {
      v16bf bk0 = *(const v16bf*)(K + (size_t)(j0 + lane15) * HEAD_DIM + h16);
      v16bf bk1 = *(const v16bf*)(K + (size_t)(j0 + lane15) * HEAD_DIM + 32 + h16);
      s0 = wmma_bf16(aq0, bk0, z0);
      s0 = wmma_bf16(aq1, bk1, s0);
      v16bf bk2 = *(const v16bf*)(K + (size_t)(j0 + 16 + lane15) * HEAD_DIM + h16);
      v16bf bk3 = *(const v16bf*)(K + (size_t)(j0 + 16 + lane15) * HEAD_DIM + 32 + h16);
      s1 = wmma_bf16(aq0, bk2, z0);
      s1 = wmma_bf16(aq1, bk3, s1);
    }

    const int key0 = j0 + lane15, key1 = j0 + 16 + lane15;
    float alpha[8];
#pragma unroll
    for (int r = 0; r < 8; ++r) {
      const int row = i0 + r + h8;
      float x0 = (key0 <= row + 1) ? s0[r] * scale : -3.0e38f;
      float x1 = (key1 <= row + 1) ? s1[r] * scale : -3.0e38f;
      s0[r] = x0; s1[r] = x1;
      float mx = fmaxf(x0, x1);
      mx = fmaxf(mx, __shfl_xor(mx, 1, 32));
      mx = fmaxf(mx, __shfl_xor(mx, 2, 32));
      mx = fmaxf(mx, __shfl_xor(mx, 4, 32));
      mx = fmaxf(mx, __shfl_xor(mx, 8, 32));
      const float mn = fmaxf(mrow[r], mx);
      alpha[r] = __expf(mrow[r] - mn);
      mrow[r]  = mn;
    }

    // P = exp(S - m); row-sum; swizzle C-layout -> A-layout in LDS (bf16).
#pragma unroll
    for (int r = 0; r < 8; ++r) {
      const int row8 = r + h8;
      const float p0 = __expf(s0[r] - mrow[r]);
      const float p1 = __expf(s1[r] - mrow[r]);
      float rs = p0 + p1;
      rs += __shfl_xor(rs, 1, 32);
      rs += __shfl_xor(rs, 2, 32);
      rs += __shfl_xor(rs, 4, 32);
      rs += __shfl_xor(rs, 8, 32);
      lrow[r] = lrow[r] * alpha[r] + rs;
      {   // k = lane15 (0..15): lane' = row8, i = k
        const int k = lane15;
        const int l2 = row8;
        pst[l2 * 16 + k] = (__bf16)p0;
      }
      {   // k = 16+lane15 (16..31): lane' = row8 + 16, i = k&15
        const int k = lane15;
        const int l2 = row8 + 16;
        pst[l2 * 16 + k] = (__bf16)p1;
      }
    }

#pragma unroll
    for (int f = 0; f < 4; ++f)
#pragma unroll
      for (int r = 0; r < 8; ++r) o[f][r] *= alpha[r];

    asm volatile("" ::: "memory");  // keep LDS store->load order at IR level
    // DS ops are in-order within a wave; each wave owns its region.
    const v16bf pa = *(const v16bf*)(pst + lane * 16);

#pragma unroll
    for (int f = 0; f < 4; ++f) {
      v16bf bv_ = *(const v16bf*)(V + (size_t)(f * 16 + lane15) * T_SEQ + j0 + h16);
      o[f] = wmma_bf16(pa, bv_, o[f]);
    }
  }

  // normalize and store ctx as bf16 [B, T, H*64] (row-major for out-proj)
#pragma unroll
  for (int r = 0; r < 8; ++r) {
    const float inv = 1.0f / lrow[r];
    const int   t   = i0 + r + h8;
#pragma unroll
    for (int f = 0; f < 4; ++f) {
      ctx[(size_t)(b * T_SEQ + t) * D_MODEL + h * HEAD_DIM + f * 16 + lane15] =
          (__bf16)(o[f][r] * inv);
    }
  }
}

// ---------------------------------------------------------------------------
// K4: out = ctx @ Wc^T + bc  (bf16 in, fp32 out)
// ---------------------------------------------------------------------------
__global__ __launch_bounds__(256) void proj_out_kernel(
    const __bf16* __restrict__ ctx, const __bf16* __restrict__ wc,
    const float* __restrict__ bc, float* __restrict__ out) {
  const int lane   = threadIdx.x & 31;
  const int wave   = threadIdx.x >> 5;
  const int lane15 = lane & 15;
  const int h8  = (lane & 16) ? 8 : 0;
  const int h16 = (lane & 16) ? 16 : 0;

  const int m_wave = blockIdx.x * 128 + (wave >> 1) * 32;
  const int n_wave = blockIdx.y * 64  + (wave & 1) * 32;

  v8f c[2][2] = {{{}, {}}, {{}, {}}};

  for (int k0 = 0; k0 < D_MODEL; k0 += 32) {
    v16bf a[2], b[2];
#pragma unroll
    for (int mi = 0; mi < 2; ++mi)
      a[mi] = load_afrag_bf16(ctx + (size_t)(m_wave + mi * 16 + lane15) * D_MODEL + k0, h8);
#pragma unroll
    for (int ni = 0; ni < 2; ++ni)
      b[ni] = *(const v16bf*)(wc + (size_t)(n_wave + ni * 16 + lane15) * D_MODEL + k0 + h16);
#pragma unroll
    for (int mi = 0; mi < 2; ++mi)
#pragma unroll
      for (int ni = 0; ni < 2; ++ni)
        c[mi][ni] = wmma_bf16(a[mi], b[ni], c[mi][ni]);
  }

#pragma unroll
  for (int ni = 0; ni < 2; ++ni) {
    const int   n  = n_wave + ni * 16 + lane15;
    const float bb = bc[n];
#pragma unroll
    for (int mi = 0; mi < 2; ++mi) {
      const int mbase = m_wave + mi * 16 + h8;
#pragma unroll
      for (int r = 0; r < 8; ++r)
        out[(size_t)(mbase + r) * D_MODEL + n] = c[mi][ni][r] + bb;
    }
  }
}

// ---------------------------------------------------------------------------
extern "C" void kernel_launch(void* const* d_in, const int* in_sizes, int n_in,
                              void* d_out, int out_size, void* d_ws, size_t ws_size,
                              hipStream_t stream) {
  const float* q  = (const float*)d_in[0];
  const float* k  = (const float*)d_in[1];
  const float* v  = (const float*)d_in[2];
  const float* Wq = (const float*)d_in[3];
  const float* bq = (const float*)d_in[4];
  const float* Wk = (const float*)d_in[5];
  const float* bk = (const float*)d_in[6];
  const float* Wv = (const float*)d_in[7];
  const float* bv = (const float*)d_in[8];
  const float* Wc = (const float*)d_in[9];
  const float* bc = (const float*)d_in[10];
  float* out = (float*)d_out;

  // workspace layout (bf16):  [4 weights][qh][kh][vT][ctx]  ~= 69 MB
  char* ws = (char*)d_ws;
  const size_t W_ELEMS  = (size_t)4 * D_MODEL * D_MODEL;
  const size_t ACT_ELEMS = (size_t)M_TOT * D_MODEL;   // 16384*512
  __bf16* wbf = (__bf16*)ws;
  __bf16* qh  = wbf + W_ELEMS;
  __bf16* kh  = qh + ACT_ELEMS;
  __bf16* vT  = kh + ACT_ELEMS;
  __bf16* ctx = vT + ACT_ELEMS;

  wcvt_kernel<<<dim3(512), dim3(256), 0, stream>>>(Wq, Wk, Wv, Wc, wbf);

  dim3 gp(M_TOT / 128, D_MODEL / 64, 3);
  proj_qkv_kernel<<<gp, dim3(256), 0, stream>>>(q, k, v, wbf, bq, bk, bv, qh, kh, vT);

  dim3 ga(T_SEQ / 128, B_SZ * N_HEAD, 1);
  attn_kernel<<<ga, dim3(256), 0, stream>>>(qh, kh, vT, ctx);

  dim3 go(M_TOT / 128, D_MODEL / 64, 1);
  proj_out_kernel<<<go, dim3(256), 0, stream>>>(ctx, wbf + (size_t)3 * D_MODEL * D_MODEL,
                                                bc, out);
}
